// SparseMoELayer_55594056680081
// MI455X (gfx1250) — matrix operand order repaired and checked by reading, not compiled, not run on hardware
//
#include <hip/hip_runtime.h>
#include <math.h>

// ---------------------------------------------------------------------------
// Sparse MoE (top-2 of 8), MI455X / gfx1250.
//   wave32 + v_wmma_f32_16x16x32_bf16 + global_load_async_to_lds_b128 (ASYNCcnt)
// B=2 N=2048 D=1024 E=8 H=2048 K=2  -> T=4096 tokens, 8192 (token,expert) rows
// ---------------------------------------------------------------------------

constexpr int Bb = 2, Nn = 2048, Dd = 1024, Ee = 8, Hh = 2048, Kk = 2;
constexpr int Tt = Bb * Nn;     // 4096 tokens
constexpr int TK = Tt * Kk;     // 8192 compacted expert rows

// workspace layout (bytes)
constexpr size_t OFF_XBF  = 0;                                 // x bf16 (T,D)
constexpr size_t SZ_XBF   = (size_t)Tt * Dd * 2;               // 8 MiB
constexpr size_t OFF_HBF  = OFF_XBF + SZ_XBF;                  // h bf16 (TK,H)
constexpr size_t SZ_HBF   = (size_t)TK * Hh * 2;               // 32 MiB
constexpr size_t OFF_W1T  = OFF_HBF + SZ_HBF;                  // w1^T bf16 (E,H,D)
constexpr size_t SZ_W1T   = (size_t)Ee * Dd * Hh * 2;          // 32 MiB
constexpr size_t OFF_W2T  = OFF_W1T + SZ_W1T;                  // w2^T bf16 (E,D,H)
constexpr size_t SZ_W2T   = (size_t)Ee * Hh * Dd * 2;          // 32 MiB
constexpr size_t OFF_IDX  = OFF_W2T + SZ_W2T;                  // token idx per (e,pos)
constexpr size_t SZ_IDX   = (size_t)Ee * Tt * 4;               // 128 KiB
constexpr size_t OFF_GV   = OFF_IDX + SZ_IDX;                  // gate val per (e,pos)
constexpr size_t SZ_GV    = (size_t)Ee * Tt * 4;               // 128 KiB
constexpr size_t OFF_CNT  = OFF_GV + SZ_GV;                    // per-expert counts
constexpr size_t OFF_OFFS = OFF_CNT + 64;                      // per-expert offsets
constexpr size_t WS_REQ   = OFF_OFFS + 64;                     // ~104.3 MB total

typedef __attribute__((ext_vector_type(16))) __bf16 v16bf;
typedef __attribute__((ext_vector_type(8)))  float  v8f;

struct AF  { uint4 lo, hi; };           // 32B -> v16bf
struct BF8 { unsigned short h[8]; };    // 16B -> uint4

__device__ __forceinline__ unsigned short f2bf(float f) {
  unsigned int u = __float_as_uint(f);
  unsigned int r = u + 0x7FFFu + ((u >> 16) & 1u);   // round-to-nearest-even
  return (unsigned short)(r >> 16);
}

// --------------------------- kernel 1: zero --------------------------------
__global__ void zero_kernel(float* __restrict__ out, int n, int* __restrict__ cnt) {
  int i = blockIdx.x * blockDim.x + threadIdx.x;
  for (int j = i; j < n; j += gridDim.x * blockDim.x) out[j] = 0.0f;
  if (i < Ee) cnt[i] = 0;
}

// --------------------------- kernel 2: x -> bf16 ---------------------------
__global__ void cvtx_kernel(const float* __restrict__ x, unsigned short* __restrict__ xbf) {
  int i = blockIdx.x * blockDim.x + threadIdx.x;     // i in [0, T*D/8)
  const float4* xv = (const float4*)x;
  float4 a = xv[2 * i], b = xv[2 * i + 1];
  BF8 o;
  o.h[0] = f2bf(a.x); o.h[1] = f2bf(a.y); o.h[2] = f2bf(a.z); o.h[3] = f2bf(a.w);
  o.h[4] = f2bf(b.x); o.h[5] = f2bf(b.y); o.h[6] = f2bf(b.z); o.h[7] = f2bf(b.w);
  ((uint4*)xbf)[i] = __builtin_bit_cast(uint4, o);
}

// ------------------- kernel 2b: transpose+convert weights ------------------
// w (E,R,C) fp32  ->  wt (E,C,R) bf16   (32x32 tiles through LDS)
__global__ __launch_bounds__(256)
void tconv_kernel(const float* __restrict__ w, unsigned short* __restrict__ wt,
                  int R, int C) {
  __shared__ float tile[32][33];
  const int e  = blockIdx.z;
  const int c0 = blockIdx.x * 32;
  const int r0 = blockIdx.y * 32;
  const int tc = threadIdx.x & 31;
  const int tr = threadIdx.x >> 5;                       // 0..7
  const float* src = w + ((size_t)e * R + r0) * C + c0;
  #pragma unroll
  for (int p = 0; p < 4; ++p)
    tile[tr + p * 8][tc] = src[(size_t)(tr + p * 8) * C + tc];
  __syncthreads();
  unsigned short* dst = wt + ((size_t)e * C + c0) * R + r0;
  #pragma unroll
  for (int p = 0; p < 4; ++p)
    dst[(size_t)(tr + p * 8) * R + tc] = f2bf(tile[tc][tr + p * 8]);
}

// --------------------------- kernel 3: gating ------------------------------
__global__ void gate_kernel(const float* __restrict__ x, const float* __restrict__ wg,
                            const float* __restrict__ bg, int* __restrict__ cnt,
                            int* __restrict__ idx, float* __restrict__ gatev,
                            float* __restrict__ entOut) {
  int t = blockIdx.x * blockDim.x + threadIdx.x;
  if (t >= Tt) return;
  float acc[Ee] = {0, 0, 0, 0, 0, 0, 0, 0};
  const float* xr = x + (size_t)t * Dd;
  for (int d = 0; d < Dd; d += 4) {
    float4 xv = *(const float4*)(xr + d);
    const float* w0 = wg + (size_t)d * Ee;
    #pragma unroll
    for (int e = 0; e < Ee; ++e)
      acc[e] += xv.x * w0[e] + xv.y * w0[Ee + e] + xv.z * w0[2 * Ee + e] + xv.w * w0[3 * Ee + e];
  }
  #pragma unroll
  for (int e = 0; e < Ee; ++e) acc[e] += bg[e];
  // top-2, first index wins ties (matches jax.lax.top_k)
  int i0 = 0; float l0 = acc[0];
  #pragma unroll
  for (int e = 1; e < Ee; ++e) if (acc[e] > l0) { l0 = acc[e]; i0 = e; }
  int i1 = (i0 == 0) ? 1 : 0; float l1 = acc[i1];
  #pragma unroll
  for (int e = 0; e < Ee; ++e) if (e != i0 && acc[e] > l1) { l1 = acc[e]; i1 = e; }
  float e1 = __expf(l1 - l0);                 // softmax over {l0, l1}, l0 >= l1
  float s  = 1.0f + e1;
  float g0 = 1.0f / s, g1 = e1 / s;
  float ent = -(g0 * __logf(fmaxf(g0, 1e-8f)) + g1 * __logf(fmaxf(g1, 1e-8f)));
  atomicAdd(entOut, ent * (1.0f / Tt));
  int p0 = atomicAdd(&cnt[i0], 1);
  idx[i0 * Tt + p0] = t; gatev[i0 * Tt + p0] = g0;
  int p1 = atomicAdd(&cnt[i1], 1);
  idx[i1 * Tt + p1] = t; gatev[i1 * Tt + p1] = g1;
}

// --------------------------- kernel 4: prefix offsets ----------------------
__global__ void offs_kernel(const int* __restrict__ cnt, int* __restrict__ offs) {
  if (blockIdx.x == 0 && threadIdx.x == 0) {
    int s = 0;
    for (int e = 0; e < Ee; ++e) { offs[e] = s; s += cnt[e]; }
  }
}

// --------------------------- GEMM core -------------------------------------
// per-lane WMMA coords (wave32):
//   A 16x32 bf16: M = lane&15; elem j -> K = 2*kA+j, kA=(lane<16)?0:8
//   B 32x16 bf16: N = lane&15; elem j -> K = kB+j, kB=(lane<16)?0:16
//   C 16x16 f32 : reg r -> M = r + ((lane<16)?0:8), N = lane&15
// Block tile M=64 x N=128, 8 waves as 2(M) x 4(N), each wave 2x2 16-tiles.
// B tiles (128 rows x 32 K, bf16, stride 40 elems = 80B, 16B aligned) are
// double-buffered in LDS and filled with GLOBAL_LOAD_ASYNC_TO_LDS_B128.

#define BSTR 40                       // LDS row stride in bf16 elements
#define BTILE (128 * BSTR)            // elements per buffer

__device__ __forceinline__ void async_tile_b128(unsigned ldsAddr, const unsigned short* g) {
  asm volatile("global_load_async_to_lds_b128 %0, %1, off" :: "v"(ldsAddr), "v"(g) : "memory");
}
__device__ __forceinline__ void wait_async0() {
  asm volatile("s_wait_asynccnt 0x0" ::: "memory");
}

__device__ __forceinline__ v16bf ld_bfrag(const unsigned short* p) {
  AF f;
  f.lo = *(const uint4*)p;
  f.hi = *(const uint4*)(p + 8);
  return __builtin_bit_cast(v16bf, f);
}
__device__ __forceinline__ v16bf ld_afrag(const unsigned short* row, int kk, int kA) {
  AF f;
  f.lo = *(const uint4*)(row + kk + 2 * kA);
  f.hi = *(const uint4*)(row + kk + 2 * kA + 8);
  return __builtin_bit_cast(v16bf, f);
}
#define WMMA_BF16(A, B, C) \
  __builtin_amdgcn_wmma_f32_16x16x32_bf16(false, (A), false, (B), (short)0, (C), false, false)

// ----- GEMM1: h[slot, :] = relu( x[token] @ w1[e] + b1[e] ), bf16 out ------
__global__ __launch_bounds__(256)
void gemm1_kernel(const unsigned short* __restrict__ xbf, const unsigned short* __restrict__ w1t,
                  const float* __restrict__ b1, const int* __restrict__ idx,
                  const int* __restrict__ cnt, const int* __restrict__ offs,
                  unsigned short* __restrict__ hbf) {
  const int e  = blockIdx.x >> 6;              // 64 M-tiles of 64 per expert
  const int m0 = (blockIdx.x & 63) * 64;
  const int count = cnt[e];
  if (m0 >= count) return;                     // uniform exit
  const int n0 = blockIdx.y * 128;

  __shared__ __align__(16) unsigned short Bt[2][BTILE];   // 2 x 10 KB

  const int tid = threadIdx.x;
  const int lane = tid & 31, wid = tid >> 5;
  const int waveM = wid & 1, waveN = wid >> 1;
  const int l16 = lane & 15, half = lane >> 4;
  const int kA = half * 8, kB = half * 16;

  // async staging: thread -> (row sn, 16-elem segment sseg) of the 128x32 tile
  const int sn = tid >> 1;
  const int sseg = (tid & 1) * 16;
  const unsigned short* grow = w1t + (size_t)(e * Hh + n0 + sn) * Dd + sseg;
  const unsigned ldsrow =
      (unsigned)(unsigned long long)&Bt[0][0] + (unsigned)(sn * BSTR + sseg) * 2u;

  // A rows (gathered tokens), clamped for partial tiles
  int mA0 = m0 + waveM * 32 + l16;      if (mA0 >= count) mA0 = m0;
  int mA1 = m0 + waveM * 32 + 16 + l16; if (mA1 >= count) mA1 = m0;
  const unsigned short* arow0 = xbf + (size_t)idx[e * Tt + mA0] * Dd;
  const unsigned short* arow1 = xbf + (size_t)idx[e * Tt + mA1] * Dd;

  v8f acc00 = {0,0,0,0,0,0,0,0}, acc01 = {0,0,0,0,0,0,0,0};
  v8f acc10 = {0,0,0,0,0,0,0,0}, acc11 = {0,0,0,0,0,0,0,0};

  constexpr int NIT = Dd / 32;
  async_tile_b128(ldsrow, grow);
  async_tile_b128(ldsrow + 16u, grow + 8);
  for (int i = 0; i < NIT; ++i) {
    const int kk = i * 32;
    wait_async0();
    __syncthreads();                    // tile i visible; all reads of buf[(i+1)&1] done
    if (i + 1 < NIT) {
      const unsigned short* g = grow + kk + 32;
      unsigned lb = ldsrow + (unsigned)(((i + 1) & 1) * BTILE * 2);
      async_tile_b128(lb, g);
      async_tile_b128(lb + 16u, g + 8);
    }
    const unsigned short* base = &Bt[i & 1][0];
    v16bf A0 = ld_afrag(arow0, kk, kA);
    v16bf A1 = ld_afrag(arow1, kk, kA);
    const unsigned short* bp0 = base + (waveN * 32 + l16) * BSTR + kB;
    v16bf B0 = ld_bfrag(bp0);
    v16bf B1 = ld_bfrag(bp0 + 16 * BSTR);
    acc00 = WMMA_BF16(A0, B0, acc00);
    acc01 = WMMA_BF16(A0, B1, acc01);
    acc10 = WMMA_BF16(A1, B0, acc10);
    acc11 = WMMA_BF16(A1, B1, acc11);
  }

  const int slotBase = offs[e];
  #pragma unroll
  for (int sub = 0; sub < 2; ++sub) {
    v8f a0 = sub ? acc10 : acc00;
    v8f a1 = sub ? acc11 : acc01;
    #pragma unroll
    for (int r = 0; r < 8; ++r) {
      int m = m0 + waveM * 32 + sub * 16 + half * 8 + r;
      if (m < count) {
        int slot = slotBase + m;
        int n_ = n0 + waveN * 32 + l16;
        float v0 = a0[r] + b1[e * Hh + n_];
        float v1 = a1[r] + b1[e * Hh + n_ + 16];
        v0 = v0 > 0.0f ? v0 : 0.0f;
        v1 = v1 > 0.0f ? v1 : 0.0f;
        hbf[(size_t)slot * Hh + n_]      = f2bf(v0);
        hbf[(size_t)slot * Hh + n_ + 16] = f2bf(v1);
      }
    }
  }
}

// ----- GEMM2: out[token] += g * ( h[slot] @ w2[e] + b2[e] ) ----------------
__global__ __launch_bounds__(256)
void gemm2_kernel(const unsigned short* __restrict__ hbf, const unsigned short* __restrict__ w2t,
                  const float* __restrict__ b2, const int* __restrict__ idx,
                  const float* __restrict__ gatev, const int* __restrict__ cnt,
                  const int* __restrict__ offs, float* __restrict__ out) {
  const int e  = blockIdx.x >> 6;
  const int m0 = (blockIdx.x & 63) * 64;
  const int count = cnt[e];
  if (m0 >= count) return;
  const int n0 = blockIdx.y * 128;

  __shared__ __align__(16) unsigned short Bt[2][BTILE];

  const int tid = threadIdx.x;
  const int lane = tid & 31, wid = tid >> 5;
  const int waveM = wid & 1, waveN = wid >> 1;
  const int l16 = lane & 15, half = lane >> 4;
  const int kA = half * 8, kB = half * 16;

  const int sn = tid >> 1;
  const int sseg = (tid & 1) * 16;
  const unsigned short* grow = w2t + (size_t)(e * Dd + n0 + sn) * Hh + sseg;
  const unsigned ldsrow =
      (unsigned)(unsigned long long)&Bt[0][0] + (unsigned)(sn * BSTR + sseg) * 2u;

  const int slotBase = offs[e];
  int mA0 = m0 + waveM * 32 + l16;      if (mA0 >= count) mA0 = m0;
  int mA1 = m0 + waveM * 32 + 16 + l16; if (mA1 >= count) mA1 = m0;
  const unsigned short* arow0 = hbf + (size_t)(slotBase + mA0) * Hh;
  const unsigned short* arow1 = hbf + (size_t)(slotBase + mA1) * Hh;

  v8f acc00 = {0,0,0,0,0,0,0,0}, acc01 = {0,0,0,0,0,0,0,0};
  v8f acc10 = {0,0,0,0,0,0,0,0}, acc11 = {0,0,0,0,0,0,0,0};

  constexpr int NIT = Hh / 32;
  async_tile_b128(ldsrow, grow);
  async_tile_b128(ldsrow + 16u, grow + 8);
  for (int i = 0; i < NIT; ++i) {
    const int kk = i * 32;
    wait_async0();
    __syncthreads();
    if (i + 1 < NIT) {
      const unsigned short* g = grow + kk + 32;
      unsigned lb = ldsrow + (unsigned)(((i + 1) & 1) * BTILE * 2);
      async_tile_b128(lb, g);
      async_tile_b128(lb + 16u, g + 8);
    }
    const unsigned short* base = &Bt[i & 1][0];
    v16bf A0 = ld_afrag(arow0, kk, kA);
    v16bf A1 = ld_afrag(arow1, kk, kA);
    const unsigned short* bp0 = base + (waveN * 32 + l16) * BSTR + kB;
    v16bf B0 = ld_bfrag(bp0);
    v16bf B1 = ld_bfrag(bp0 + 16 * BSTR);
    acc00 = WMMA_BF16(A0, B0, acc00);
    acc01 = WMMA_BF16(A0, B1, acc01);
    acc10 = WMMA_BF16(A1, B0, acc10);
    acc11 = WMMA_BF16(A1, B1, acc11);
  }

  #pragma unroll
  for (int sub = 0; sub < 2; ++sub) {
    v8f a0 = sub ? acc10 : acc00;
    v8f a1 = sub ? acc11 : acc01;
    #pragma unroll
    for (int r = 0; r < 8; ++r) {
      int m = m0 + waveM * 32 + sub * 16 + half * 8 + r;
      if (m < count) {
        int token = idx[e * Tt + m];
        float g   = gatev[e * Tt + m];
        int n_ = n0 + waveN * 32 + l16;
        float v0 = (a0[r] + b2[e * Dd + n_]) * g;
        float v1 = (a1[r] + b2[e * Dd + n_ + 16]) * g;
        atomicAdd(&out[(size_t)token * Dd + n_], v0);
        atomicAdd(&out[(size_t)token * Dd + n_ + 16], v1);
      }
    }
  }
}

// ---------------------------------------------------------------------------
extern "C" void kernel_launch(void* const* d_in, const int* in_sizes, int n_in,
                              void* d_out, int out_size, void* d_ws, size_t ws_size,
                              hipStream_t stream) {
  const float* x  = (const float*)d_in[0];
  const float* wg = (const float*)d_in[1];
  const float* bg = (const float*)d_in[2];
  const float* w1 = (const float*)d_in[3];
  const float* b1 = (const float*)d_in[4];
  const float* w2 = (const float*)d_in[5];
  const float* b2 = (const float*)d_in[6];
  float* out = (float*)d_out;                 // T*D floats, then entropy at [T*D]

  if (ws_size < WS_REQ) return;               // need ~104.3 MB scratch
  char* ws = (char*)d_ws;
  unsigned short* xbf  = (unsigned short*)(ws + OFF_XBF);
  unsigned short* hbf  = (unsigned short*)(ws + OFF_HBF);
  unsigned short* w1t  = (unsigned short*)(ws + OFF_W1T);
  unsigned short* w2t  = (unsigned short*)(ws + OFF_W2T);
  int*            idx  = (int*)(ws + OFF_IDX);
  float*          gv   = (float*)(ws + OFF_GV);
  int*            cnt  = (int*)(ws + OFF_CNT);
  int*            offs = (int*)(ws + OFF_OFFS);

  // 1) zero output (atomically accumulated) + entropy slot + expert counters
  zero_kernel<<<2048, 256, 0, stream>>>(out, Tt * Dd + 1, cnt);
  // 2) x -> bf16; w1,w2 -> transposed bf16 (K-contiguous for WMMA B operand)
  cvtx_kernel<<<2048, 256, 0, stream>>>(x, xbf);
  tconv_kernel<<<dim3(Hh / 32, Dd / 32, Ee), 256, 0, stream>>>(w1, w1t, Dd, Hh);
  tconv_kernel<<<dim3(Dd / 32, Hh / 32, Ee), 256, 0, stream>>>(w2, w2t, Hh, Dd);
  // 3) routing: logits, top-2, softmax, entropy, per-expert token lists
  gate_kernel<<<Tt / 256, 256, 0, stream>>>(x, wg, bg, cnt, idx, gv, out + (size_t)Tt * Dd);
  // 4) per-expert offsets into compacted row space
  offs_kernel<<<1, 1, 0, stream>>>(cnt, offs);
  // 5) h = relu(Xg @ W1 + b1)   grid: (E * T/64 M-tiles, H/128 N-tiles)
  gemm1_kernel<<<dim3(Ee * (Tt / 64), Hh / 128), 256, 0, stream>>>(
      xbf, w1t, b1, idx, cnt, offs, hbf);
  // 6) out += g * (h @ W2 + b2) grid: (E * T/64 M-tiles, D/128 N-tiles)
  gemm2_kernel<<<dim3(Ee * (Tt / 64), Dd / 128), 256, 0, stream>>>(
      hbf, w2t, b2, idx, gv, cnt, offs, out);
}
